// ScaledDotProductAttention_42855183679993
// MI455X (gfx1250) — compile-verified, matrix operand order checked by reading, not run on hardware
//
#include <hip/hip_runtime.h>
#include <cstdint>
#include <cstddef>

typedef __attribute__((ext_vector_type(16))) _Float16 v16h;
typedef __attribute__((ext_vector_type(8)))  _Float16 h8;
typedef __attribute__((ext_vector_type(2)))  _Float16 h2;
typedef __attribute__((ext_vector_type(8)))  float    v8f;
typedef __attribute__((ext_vector_type(4)))  unsigned int u32x4;
typedef __attribute__((ext_vector_type(8)))  int      i32x8;
typedef __attribute__((ext_vector_type(4)))  int      i32x4;

namespace {
constexpr int   kB          = 16;
constexpr int   kL          = 2048;
constexpr int   kD          = 64;
constexpr int   kKB         = 32;     // keys per staged tile
constexpr int   kQRowsPerWG = 128;    // 8 waves x 16 query rows
constexpr int   kT          = kL / kKB;          // 64 tiles
// softmax kept in the exp2 domain: score2 = (qk/8)*log2(e)
constexpr float kScale2     = 0.125f * 1.44269504088896340736f;
constexpr float kNeg2       = -1000.0f * 1.44269504088896340736f;
}

// v_exp_f32 directly (no libm wrapper).
__device__ __forceinline__ float exp2_fast(float x) {
  return __builtin_amdgcn_exp2f(x);
}

// Per lane-half, a 16-element f16 fragment is two contiguous 8-elem runs:
// K indices {8h..8h+7} and {16+8h..16+8h+7}  ->  two ds_load_b128.
__device__ __forceinline__ v16h load_frag(const _Float16* row, int half) {
  const h8 lo = *(const h8*)(row + 8 * half);
  const h8 hi = *(const h8*)(row + 16 + 8 * half);
  v16h f;
  #pragma unroll
  for (int i = 0; i < 8; ++i) { f[i] = lo[i]; f[i + 8] = hi[i]; }
  return f;
}

// Reductions across a 16-lane half (rows r / r+8 live in separate halves).
__device__ __forceinline__ float halfmax(float x) {
  #pragma unroll
  for (int m = 1; m < 16; m <<= 1) x = fmaxf(x, __shfl_xor(x, m, 32));
  return x;
}
__device__ __forceinline__ float halfsum(float x) {
  #pragma unroll
  for (int m = 1; m < 16; m <<= 1) x += __shfl_xor(x, m, 32);
  return x;
}

// TDM load of a [kKB x kD] f32 tile (row-major, row stride kD) into LDS.
// D# per CDNA5 ISA sec.8; 6-arg builtin on this toolchain.
__device__ __forceinline__ void tdm_load_tile(const float* gbase, unsigned lds_off) {
  const unsigned long long ga = (unsigned long long)(uintptr_t)gbase;
  u32x4 g0;
  g0[0] = 1u;                                   // count=1 (valid descriptor)
  g0[1] = lds_off;                              // lds_addr
  g0[2] = (unsigned)ga;                         // global_addr[31:0]
  g0[3] = (unsigned)((ga >> 32) & 0x01FFFFFFu)  // global_addr[56:32]
        | (2u << 30);                           // type=2 ("image")
  i32x8 g1;
  g1[0] = (2 << 16);                 // data_size=2 (4B), workgroup_mask=0
  g1[1] = (kD & 0xFFFF) << 16;       // tensor_dim0 = 64
  g1[2] = (kL & 0xFFFF) << 16;       // tensor_dim1 = 2048
  g1[3] = (kD << 16);                // tile_dim0 = 64
  g1[4] = kKB;                       // tile_dim1 = 32
  g1[5] = kD;                        // tensor_dim0_stride = 64
  g1[6] = 0;
  g1[7] = 0;
  i32x4 z4 = {0, 0, 0, 0};
  i32x8 z8 = {0, 0, 0, 0, 0, 0, 0, 0};
  __builtin_amdgcn_tensor_load_to_lds(g0, g1, z4, z4, z8, 0);
}

// Cooperative f32->f16 convert, row-major (K tile): 1024 packed pairs.
__device__ __forceinline__ void convert_rows(const float* src, _Float16* dst, int tid) {
  #pragma unroll
  for (int idx = tid; idx < (kKB * kD) / 2; idx += 256) {
    const float2 f = ((const float2*)src)[idx];
    h2 p; p[0] = (_Float16)f.x; p[1] = (_Float16)f.y;
    ((h2*)dst)[idx] = p;
  }
}

// Cooperative f32->f16 convert + transpose (V tile): dst[dim][key].
__device__ __forceinline__ void convert_vt(const float* src, _Float16* dst, int tid) {
  #pragma unroll
  for (int idx = tid; idx < (kKB * kD) / 2; idx += 256) {
    const int d = idx >> 4;            // kKB/2 = 16 packed key-pairs per dim
    const int c = idx & 15;
    h2 p;
    p[0] = (_Float16)src[(2 * c)     * kD + d];
    p[1] = (_Float16)src[(2 * c + 1) * kD + d];
    ((h2*)dst)[d * (kKB / 2) + c] = p;
  }
}

#define WMMA_F16(A, Bm, C) \
  __builtin_amdgcn_wmma_f32_16x16x32_f16(false, (A), false, (Bm), (short)0, (C), false, false)

__global__ __launch_bounds__(256)
void sdpa_wmma_kernel(const float* __restrict__ q, const float* __restrict__ k,
                      const float* __restrict__ v, const unsigned char* __restrict__ pad,
                      float* __restrict__ out, float* __restrict__ attn) {
  __shared__ __align__(16) float    sK32[2][kKB * kD];   // TDM landing, f32
  __shared__ __align__(16) float    sV32[2][kKB * kD];
  __shared__ __align__(16) _Float16 sKh[2][kKB * kD];    // f16 [key][dim]
  __shared__ __align__(16) _Float16 sVt[2][kD * kKB];    // f16 [dim][key]
  __shared__ __align__(16) _Float16 sP[8][16 * kKB];     // per-wave P tile

  const int tid  = threadIdx.x;
  const int lane = tid & 31;
  const int wave = tid >> 5;
  const int half = lane >> 4;
  const int l16  = lane & 15;

  const int blocksPerBatch = kL / kQRowsPerWG;  // 16
  const int b  = blockIdx.x / blocksPerBatch;
  const int qb = (blockIdx.x % blocksPerBatch) * kQRowsPerWG + wave * 16;

  const float* qB = q + (size_t)b * kL * kD;
  const float* kB = k + (size_t)b * kL * kD;
  const float* vB = v + (size_t)b * kL * kD;
  const unsigned char* padB = pad + (size_t)b * kL;

  // ---- Q A-fragments (16x64 f16, K split 0..31 / 32..63), converted once ----
  v16h aq0, aq1;
  {
    const float* qr = qB + (size_t)(qb + l16) * kD;
    #pragma unroll
    for (int i = 0; i < 8; ++i) {
      aq0[i]     = (_Float16)qr[8 * half + i];
      aq0[i + 8] = (_Float16)qr[16 + 8 * half + i];
      aq1[i]     = (_Float16)qr[32 + 8 * half + i];
      aq1[i + 8] = (_Float16)qr[48 + 8 * half + i];
    }
  }

  // ================= Pass 1: row max + rescaled row sum (exp2 domain) ======
  float m[8], lsum[8];
  #pragma unroll
  for (int r = 0; r < 8; ++r) { m[r] = -3.0e38f; lsum[r] = 0.0f; }

  if (wave == 0) {
    tdm_load_tile(kB, (unsigned)(uintptr_t)&sK32[0][0]);
    __builtin_amdgcn_s_wait_tensorcnt(0);
  }
  __syncthreads();

  for (int t = 0; t < kT; ++t) {
    const int cur = t & 1, nxt = cur ^ 1;
    convert_rows(&sK32[cur][0], &sKh[cur][0], tid);
    __syncthreads();

    if (t + 1 < kT) {
      if (wave == 0)  // async: overlap next tile's DMA with this tile's math
        tdm_load_tile(kB + (size_t)(t + 1) * kKB * kD, (unsigned)(uintptr_t)&sK32[nxt][0]);
      if (t + 2 < kT)  // warm L2 one tile further ahead
        __builtin_prefetch(kB + (size_t)(t + 2) * kKB * kD + tid * 8, 0, 1);
    }

    const bool pad0 = padB[t * kKB + l16] != 0;
    const bool pad1 = padB[t * kKB + 16 + l16] != 0;

    const _Float16* r0 = &sKh[cur][(size_t)l16 * kD];
    const _Float16* r1 = &sKh[cur][(size_t)(l16 + 16) * kD];
    v8f s0 = {}, s1 = {};
    s0 = WMMA_F16(aq0, load_frag(r0, half), s0);
    s0 = WMMA_F16(aq1, load_frag(r0 + 32, half), s0);
    s1 = WMMA_F16(aq0, load_frag(r1, half), s1);
    s1 = WMMA_F16(aq1, load_frag(r1 + 32, half), s1);

    #pragma unroll
    for (int r = 0; r < 8; ++r) {
      const float a = pad0 ? kNeg2 : s0[r] * kScale2;
      const float c = pad1 ? kNeg2 : s1[r] * kScale2;
      const float mn = fmaxf(m[r], halfmax(fmaxf(a, c)));
      const float rs = halfsum(exp2_fast(a - mn) + exp2_fast(c - mn));
      lsum[r] = lsum[r] * exp2_fast(m[r] - mn) + rs;
      m[r] = mn;
    }

    if (t + 1 < kT && wave == 0) __builtin_amdgcn_s_wait_tensorcnt(0);
    __syncthreads();
  }

  float invl[8];
  #pragma unroll
  for (int r = 0; r < 8; ++r) invl[r] = 1.0f / lsum[r];

  // ====== Pass 2: recompute S, write normalized attn once, O += P*V ========
  v8f o[4] = {};
  _Float16* pw = &sP[wave][0];

  if (wave == 0) {
    tdm_load_tile(kB, (unsigned)(uintptr_t)&sK32[0][0]);
    tdm_load_tile(vB, (unsigned)(uintptr_t)&sV32[0][0]);
    __builtin_amdgcn_s_wait_tensorcnt(0);
  }
  __syncthreads();

  for (int t = 0; t < kT; ++t) {
    const int cur = t & 1, nxt = cur ^ 1;
    convert_rows(&sK32[cur][0], &sKh[cur][0], tid);
    convert_vt(&sV32[cur][0], &sVt[cur][0], tid);
    __syncthreads();

    if (t + 1 < kT && wave == 0) {
      tdm_load_tile(kB + (size_t)(t + 1) * kKB * kD, (unsigned)(uintptr_t)&sK32[nxt][0]);
      tdm_load_tile(vB + (size_t)(t + 1) * kKB * kD, (unsigned)(uintptr_t)&sV32[nxt][0]);
    }

    const bool pad0 = padB[t * kKB + l16] != 0;
    const bool pad1 = padB[t * kKB + 16 + l16] != 0;

    const _Float16* r0 = &sKh[cur][(size_t)l16 * kD];
    const _Float16* r1 = &sKh[cur][(size_t)(l16 + 16) * kD];
    v8f s0 = {}, s1 = {};
    s0 = WMMA_F16(aq0, load_frag(r0, half), s0);
    s0 = WMMA_F16(aq1, load_frag(r0 + 32, half), s0);
    s1 = WMMA_F16(aq0, load_frag(r1, half), s1);
    s1 = WMMA_F16(aq1, load_frag(r1 + 32, half), s1);

    // Normalized probabilities -> global attn (coalesced b32) + LDS P tile.
    #pragma unroll
    for (int r = 0; r < 8; ++r) {
      const float a = pad0 ? kNeg2 : s0[r] * kScale2;
      const float c = pad1 ? kNeg2 : s1[r] * kScale2;
      const float p0 = exp2_fast(a - m[r]) * invl[r];
      const float p1 = exp2_fast(c - m[r]) * invl[r];
      const int row = r + 8 * half;
      const size_t arow = ((size_t)b * kL + (qb + row)) * kL + t * kKB;
      attn[arow + l16]      = p0;
      attn[arow + 16 + l16] = p1;
      pw[row * kKB + l16]      = (_Float16)p0;
      pw[row * kKB + 16 + l16] = (_Float16)p1;
    }
    asm volatile("s_wait_dscnt 0" ::: "memory");  // intra-wave LDS store->load order

    // P A-fragment: row-major P tile -> two b128 chunks per lane.
    const v16h ap = load_frag(pw + l16 * kKB, half);

    // O[16x64] += P[16x32] x V[32x64]; Vt rows give b128 fragment chunks.
    #pragma unroll
    for (int n = 0; n < 4; ++n)
      o[n] = WMMA_F16(ap, load_frag(&sVt[cur][(n * 16 + l16) * kKB], half), o[n]);

    if (t + 1 < kT && wave == 0) __builtin_amdgcn_s_wait_tensorcnt(0);
    __syncthreads();
  }

  // ---- write O (already normalized) ----
  #pragma unroll
  for (int n = 0; n < 4; ++n) {
    #pragma unroll
    for (int r = 0; r < 8; ++r) {
      const int row = qb + r + 8 * half;
      out[((size_t)b * kL + row) * kD + n * 16 + l16] = o[n][r];
    }
  }
}

extern "C" void kernel_launch(void* const* d_in, const int* in_sizes, int n_in,
                              void* d_out, int out_size, void* d_ws, size_t ws_size,
                              hipStream_t stream) {
  (void)in_sizes; (void)n_in; (void)out_size; (void)d_ws; (void)ws_size;
  const float* q = (const float*)d_in[0];
  const float* k = (const float*)d_in[1];
  const float* v = (const float*)d_in[2];
  const unsigned char* pad = (const unsigned char*)d_in[3];
  float* out  = (float*)d_out;                       // [16,2048,64]
  float* attn = out + (size_t)kB * kL * kD;          // [16,2048,2048]

  dim3 grid(kB * (kL / kQRowsPerWG));  // 256 workgroups
  dim3 block(256);                     // 8 wave32 per workgroup
  sdpa_wmma_kernel<<<grid, block, 0, stream>>>(q, k, v, pad, out, attn);
}
